// RelativePositionAttention_1271310320138
// MI455X (gfx1250) — matrix operand order repaired and checked by reading, not compiled, not run on hardware
//
#include <hip/hip_runtime.h>
#include <math.h>

// Relative-position attention == flash attention over K_eff = scale*K + rel_k,
// V_eff = V + rel_v.
//   Pass 1: K_eff^T / V_eff -> bf16 workspace (one-time transform, ~16MB).
//   Pass 2: flash attention; bf16 tiles DMA'd into LDS by the Tensor Data Mover
//           (tensor_load_to_lds, TENSORcnt, double-buffered), WMMA bf16 GEMMs,
//           online softmax, f32 accumulate/output.

#define B_  2
#define S_  2048
#define D_  1024
#define H_  16
#define HD_ 64
#define LP  72   // LDS row stride (bf16 elems): 64 data + 8 pad = 144B, 16B aligned
#define NT  (S_ / 64)

typedef __attribute__((ext_vector_type(16))) __bf16       v16bf;
typedef __attribute__((ext_vector_type(8)))  float        v8f;
typedef __attribute__((ext_vector_type(4)))  unsigned int u32x4;
typedef __attribute__((ext_vector_type(2)))  unsigned int u32x2;
typedef __attribute__((ext_vector_type(8)))  int          i32x8;
typedef __attribute__((ext_vector_type(4)))  int          i32x4;

#ifndef __has_builtin
#define __has_builtin(x) 0
#endif
#if __has_builtin(__builtin_amdgcn_tensor_load_to_lds) && \
    __has_builtin(__builtin_amdgcn_s_wait_tensorcnt)
#define HAVE_TDM 1
#else
#define HAVE_TDM 0
#endif

union Frag16 {
    v16bf        v;
    u32x4        q[2];
    unsigned int u[8];
};

__device__ __forceinline__ unsigned short f2bf(float f) {
    union { float f; unsigned int u; } x; x.f = f;
    unsigned int r = x.u + 0x7FFFu + ((x.u >> 16) & 1u);   // round-to-nearest-even
    return (unsigned short)(r >> 16);
}
__device__ __forceinline__ unsigned int pack2bf(float lo, float hi) {
    return (unsigned int)f2bf(lo) | ((unsigned int)f2bf(hi) << 16);
}

#if HAVE_TDM
// One 2D TDM descriptor: 64x64 bf16 tile, global row stride `stride0` elements,
// LDS destination padded +4 dwords after every 32 dwords (=> LP=72 row stride).
__device__ __forceinline__ void tdm_load_tile(unsigned lds_addr, const unsigned short* gptr,
                                              unsigned tensor_d0, unsigned tensor_d1,
                                              unsigned stride0) {
    unsigned long long ga = (unsigned long long)(uintptr_t)gptr;
    u32x4 g0;
    g0[0] = 1u;                                          // count=1, user descriptor
    g0[1] = lds_addr;                                    // LDS byte address
    g0[2] = (unsigned)(ga & 0xFFFFFFFFu);                // global_addr[31:0]
    g0[3] = (unsigned)((ga >> 32) & 0x1FFFFFFu) | (2u << 30);  // addr[56:32] | type=2
    i32x8 g1;
    g1[0] = (int)((1u << 16)      // data_size = 2 bytes (bf16)
                | (1u << 20)      // pad_enable
                | (4u << 22)      // pad_interval: every 32 dwords (128B row)
                | (3u << 25));    // pad_amount: 4 dwords (16B)
    g1[1] = (int)((tensor_d0 & 0xFFFFu) << 16);                               // dim0 lo
    g1[2] = (int)(((tensor_d0 >> 16) & 0xFFFFu) | ((tensor_d1 & 0xFFFFu) << 16));
    g1[3] = (int)(((tensor_d1 >> 16) & 0xFFFFu) | (64u << 16));               // tile_dim0=64
    g1[4] = (int)64;                                                          // tile_dim1=64
    g1[5] = (int)stride0;                                                     // dim0 stride lo
    g1[6] = 0;
    g1[7] = 0;
    i32x4 z4 = {0, 0, 0, 0};
#if __clang_major__ >= 23
    i32x8 z8 = {0, 0, 0, 0, 0, 0, 0, 0};
    __builtin_amdgcn_tensor_load_to_lds(g0, g1, z4, z4, z8, 0);
#else
    __builtin_amdgcn_tensor_load_to_lds(g0, g1, z4, z4, 0);
#endif
}
#endif  // HAVE_TDM

// ---------------- pass 1: build bf16 K_eff^T and V_eff in workspace ----------------
__global__ __launch_bounds__(128)
void rpa_pre_kernel(const float* __restrict__ K, const float* __restrict__ V,
                    const float* __restrict__ RK, const float* __restrict__ RV,
                    unsigned short* __restrict__ KT,   // [B][H][HD][S] bf16
                    unsigned short* __restrict__ VW) { // [B][H][S][HD] bf16
    __shared__ __align__(16) unsigned short t[64][LP];
    const int tid  = threadIdx.x;
    const int h    = blockIdx.y;
    const int b    = blockIdx.z;
    const int key0 = blockIdx.x * 64;
    const int kk   = tid >> 1;
    const int dseg = (tid & 1) * 32;
    const float scale = 0.125f;

    const size_t gofs = ((size_t)(b * S_ + key0 + kk)) * D_ + h * HD_ + dseg;
    const size_t rofs = (size_t)(key0 + kk) * HD_ + dseg;

    const float* kp  = K  + gofs;
    const float* rkp = RK + rofs;
#pragma unroll
    for (int j = 0; j < 8; ++j) {
        float4 kv = *(const float4*)(kp  + 4 * j);
        float4 rk = *(const float4*)(rkp + 4 * j);
        int d = dseg + 4 * j;
        t[d + 0][kk] = f2bf(kv.x * scale + rk.x);
        t[d + 1][kk] = f2bf(kv.y * scale + rk.y);
        t[d + 2][kk] = f2bf(kv.z * scale + rk.z);
        t[d + 3][kk] = f2bf(kv.w * scale + rk.w);
    }
    const float* vp  = V  + gofs;
    const float* rvp = RV + rofs;
    unsigned short* vout = VW + ((size_t)((b * H_ + h) * S_ + key0 + kk)) * HD_ + dseg;
#pragma unroll
    for (int j = 0; j < 8; ++j) {
        float4 vv = *(const float4*)(vp  + 4 * j);
        float4 rv = *(const float4*)(rvp + 4 * j);
        u32x2 p;
        p[0] = pack2bf(vv.x + rv.x, vv.y + rv.y);
        p[1] = pack2bf(vv.z + rv.z, vv.w + rv.w);
        *(u32x2*)(vout + 4 * j) = p;
    }
    __syncthreads();
    // write K_eff^T tile out, coalesced 64B chunks
    const int r  = tid >> 1;
    const int c0 = (tid & 1) * 32;
    unsigned short* kout = KT + ((size_t)((b * H_ + h) * HD_ + r)) * S_ + key0 + c0;
#pragma unroll
    for (int j = 0; j < 4; ++j)
        *(u32x4*)(kout + 8 * j) = *(const u32x4*)&t[r][c0 + 8 * j];
}

// ---------------- pass 2: flash attention ----------------
template <bool USE_WS>
__global__ __launch_bounds__(128)
void rpa_flash_kernel(const float* __restrict__ Q, const float* __restrict__ K,
                      const float* __restrict__ V, const float* __restrict__ RK,
                      const float* __restrict__ RV,
                      const unsigned short* __restrict__ KT,
                      const unsigned short* __restrict__ VW,
                      float* __restrict__ O) {
    __shared__ __align__(16) unsigned short sKT[2][64][LP];  // K_eff^T : [d][key]
    __shared__ __align__(16) unsigned short sV [2][64][LP];  // V_eff   : [key][d]
    __shared__ __align__(16) unsigned short sP [4][16][LP];  // per-wave P staging

    const int tid  = threadIdx.x;
    const int lane = tid & 31;
    const int wave = tid >> 5;
    const int row  = lane & 15;
    const int half = lane >> 4;

    const int h  = blockIdx.y;
    const int b  = blockIdx.z;
    const int q0 = blockIdx.x * 64 + wave * 16;
    const float scale = 0.125f;

    const unsigned short* ktBase = KT + ((size_t)(b * H_ + h) * HD_) * S_;  // [d][key]
    const unsigned short* vwBase = VW + ((size_t)(b * H_ + h) * S_) * HD_;  // [key][d]

    // ---- Q A-fragments (16x32 bf16, two d-chunks) ----
    Frag16 qa[2];
    {
        const float* qrow = Q + ((size_t)(b * S_ + q0 + row)) * D_ + h * HD_;
#pragma unroll
        for (int f = 0; f < 2; ++f) {
#pragma unroll
            for (int g = 0; g < 8; ++g) {
                int d = f * 32 + ((g & 4) ? 16 : 0) + half * 8 + 2 * (g & 3);
                float2 v2 = *(const float2*)(qrow + d);
                qa[f].u[g] = pack2bf(v2.x, v2.y);
            }
        }
    }

    v8f   acc[4];
    float mrun[8], lrun[8];
#pragma unroll
    for (int n = 0; n < 4; ++n) acc[n] = (v8f){0, 0, 0, 0, 0, 0, 0, 0};
#pragma unroll
    for (int g = 0; g < 8; ++g) { mrun[g] = -1.0e30f; lrun[g] = 0.0f; }

#if HAVE_TDM
    if constexpr (USE_WS) {
        if (wave == 0) {   // prologue: DMA tile 0 into buffer 0
            tdm_load_tile((unsigned)(uintptr_t)&sKT[0][0][0], ktBase, S_, HD_, S_);
            tdm_load_tile((unsigned)(uintptr_t)&sV [0][0][0], vwBase, HD_, S_, HD_);
        }
    }
#endif

    for (int kt = 0; kt < NT; ++kt) {
        const int cur  = kt & 1;
        const int key0 = kt * 64;

        // ---- stage tile kt ----
        if constexpr (USE_WS) {
#if HAVE_TDM
            if (wave == 0) {
                if (kt + 1 < NT) {   // prefetch next tile into the other buffer
                    const int k1 = (kt + 1) * 64;
                    tdm_load_tile((unsigned)(uintptr_t)&sKT[cur ^ 1][0][0],
                                  ktBase + k1, S_, HD_, S_);
                    tdm_load_tile((unsigned)(uintptr_t)&sV[cur ^ 1][0][0],
                                  vwBase + (size_t)k1 * HD_, HD_, S_, HD_);
                    __builtin_amdgcn_s_wait_tensorcnt((short)2);  // tile kt done (in-order)
                } else {
                    __builtin_amdgcn_s_wait_tensorcnt((short)0);
                }
            }
            __syncthreads();
#else
            __syncthreads();   // prior readers of this buffer are done
#pragma unroll
            for (int i = 0; i < 4; ++i) {
                int idx = tid + i * 128;       // 512 16B-chunks per tile
                int r = idx >> 3;
                int c = (idx & 7) * 8;
                *(u32x4*)&sKT[cur][r][c] =
                    *(const u32x4*)(ktBase + (size_t)r * S_ + key0 + c);
                *(u32x4*)&sV[cur][r][c] =
                    *(const u32x4*)(vwBase + (size_t)(key0 + r) * HD_ + c);
            }
            __syncthreads();
#endif
        } else {
            __syncthreads();
            {   // stage from f32 inputs with fused transform (self-contained path)
                const int kk   = tid >> 1;
                const int dseg = (tid & 1) * 32;
                const size_t gofs = ((size_t)(b * S_ + key0 + kk)) * D_ + h * HD_ + dseg;
                const size_t rofs = (size_t)(key0 + kk) * HD_ + dseg;
                const float* kp  = K  + gofs;
                const float* rkp = RK + rofs;
#pragma unroll
                for (int j = 0; j < 8; ++j) {
                    float4 kv = *(const float4*)(kp  + 4 * j);
                    float4 rk = *(const float4*)(rkp + 4 * j);
                    int d = dseg + 4 * j;
                    sKT[cur][d + 0][kk] = f2bf(kv.x * scale + rk.x);
                    sKT[cur][d + 1][kk] = f2bf(kv.y * scale + rk.y);
                    sKT[cur][d + 2][kk] = f2bf(kv.z * scale + rk.z);
                    sKT[cur][d + 3][kk] = f2bf(kv.w * scale + rk.w);
                }
                const float* vp  = V  + gofs;
                const float* rvp = RV + rofs;
#pragma unroll
                for (int j = 0; j < 8; ++j) {
                    float4 vv = *(const float4*)(vp  + 4 * j);
                    float4 rv = *(const float4*)(rvp + 4 * j);
                    int d = dseg + 4 * j;
                    *(unsigned int*)&sV[cur][kk][d]     = pack2bf(vv.x + rv.x, vv.y + rv.y);
                    *(unsigned int*)&sV[cur][kk][d + 2] = pack2bf(vv.z + rv.z, vv.w + rv.w);
                }
            }
            __syncthreads();
        }

        // ---- S = Q @ K_eff^T : 16x64 scores per wave (8 WMMAs) ----
        v8f sc[4];
#pragma unroll
        for (int n = 0; n < 4; ++n) {
            Frag16 kb0, kb1;                  // B-frags: lane = key, vgpr pair = d
#pragma unroll
            for (int g = 0; g < 8; ++g) {
                int k = ((g & 4) ? 16 : 0) + half * 8 + 2 * (g & 3);
                kb0.u[g] = (unsigned int)sKT[cur][k][n * 16 + row] | ((unsigned int)sKT[cur][k + 1][n * 16 + row] << 16);
                kb1.u[g] = (unsigned int)sKT[cur][32 + k][n * 16 + row] | ((unsigned int)sKT[cur][33 + k][n * 16 + row] << 16);
            }
            v8f c = (v8f){0, 0, 0, 0, 0, 0, 0, 0};
            c = __builtin_amdgcn_wmma_f32_16x16x32_bf16(false, qa[0].v, false, kb0.v,
                                                        (short)0, c, false, false);
            c = __builtin_amdgcn_wmma_f32_16x16x32_bf16(false, qa[1].v, false, kb1.v,
                                                        (short)0, c, false, false);
            sc[n] = c;
        }

        // ---- online softmax ----
#pragma unroll
        for (int g = 0; g < 8; ++g) {
            float m = fmaxf(fmaxf(sc[0][g], sc[1][g]), fmaxf(sc[2][g], sc[3][g]));
#pragma unroll
            for (int off = 8; off >= 1; off >>= 1)
                m = fmaxf(m, __shfl_xor(m, off, 32));
            float mnew = fmaxf(mrun[g], m);
            float corr = __expf(mrun[g] - mnew);
            mrun[g] = mnew;
            float rs = 0.0f;
#pragma unroll
            for (int n = 0; n < 4; ++n) {
                float p = __expf(sc[n][g] - mnew);
                sc[n][g] = p;
                rs += p;
            }
#pragma unroll
            for (int off = 8; off >= 1; off >>= 1)
                rs += __shfl_xor(rs, off, 32);
            lrun[g] = lrun[g] * corr + rs;
#pragma unroll
            for (int n = 0; n < 4; ++n) acc[n][g] *= corr;
        }

        // ---- C-layout P -> bf16 LDS -> A-layout fragments (wave-private) ----
#pragma unroll
        for (int n = 0; n < 4; ++n)
#pragma unroll
            for (int g = 0; g < 8; ++g)
                sP[wave][g + half * 8][n * 16 + row] = f2bf(sc[n][g]);

        Frag16 pa0, pa1;
        {
            const unsigned short* pr = &sP[wave][row][0];
            pa0.q[0] = *(const u32x4*)(pr + half * 8);
            pa0.q[1] = *(const u32x4*)(pr + 16 + half * 8);
            pa1.q[0] = *(const u32x4*)(pr + 32 + half * 8);
            pa1.q[1] = *(const u32x4*)(pr + 48 + half * 8);
        }

        // ---- acc += P @ V_eff (8 WMMAs) ----
#pragma unroll
        for (int n = 0; n < 4; ++n) {
            Frag16 vb0, vb1;                  // B-frags: lane = d, vgpr pair = key
#pragma unroll
            for (int g = 0; g < 8; ++g) {
                int k = ((g & 4) ? 16 : 0) + half * 8 + 2 * (g & 3);
                vb0.u[g] = (unsigned int)sV[cur][k][n * 16 + row] | ((unsigned int)sV[cur][k + 1][n * 16 + row] << 16);
                vb1.u[g] = (unsigned int)sV[cur][32 + k][n * 16 + row] | ((unsigned int)sV[cur][33 + k][n * 16 + row] << 16);
            }
            acc[n] = __builtin_amdgcn_wmma_f32_16x16x32_bf16(false, pa0.v, false, vb0.v,
                                                             (short)0, acc[n], false, false);
            acc[n] = __builtin_amdgcn_wmma_f32_16x16x32_bf16(false, pa1.v, false, vb1.v,
                                                             (short)0, acc[n], false, false);
        }

        __syncthreads();   // all readers of buf[cur] done; safe to refill next iter
    }

    // ---- epilogue: normalize, store fp32 ----
#pragma unroll
    for (int g = 0; g < 8; ++g) {
        float inv = 1.0f / lrun[g];
        float* op = O + ((size_t)(b * S_ + q0 + g + half * 8)) * D_ + h * HD_;
#pragma unroll
        for (int n = 0; n < 4; ++n)
            op[n * 16 + row] = acc[n][g] * inv;
    }
}

extern "C" void kernel_launch(void* const* d_in, const int* in_sizes, int n_in,
                              void* d_out, int out_size, void* d_ws, size_t ws_size,
                              hipStream_t stream) {
    (void)in_sizes; (void)n_in; (void)out_size;
    const float* q  = (const float*)d_in[0];
    const float* k  = (const float*)d_in[1];
    const float* v  = (const float*)d_in[2];
    const float* rk = (const float*)d_in[3];
    const float* rv = (const float*)d_in[4];
    float* out = (float*)d_out;

    const size_t elems = (size_t)B_ * H_ * S_ * HD_;          // 4M elements each
    const size_t need  = 2 * elems * sizeof(unsigned short);  // K_eff^T + V_eff = 16MB

    dim3 grid(S_ / 64, H_, B_);
    dim3 block(128);

    if (ws_size >= need) {
        unsigned short* KT = (unsigned short*)d_ws;
        unsigned short* VW = KT + elems;
        rpa_pre_kernel<<<grid, block, 0, stream>>>(k, v, rk, rv, KT, VW);
        rpa_flash_kernel<true><<<grid, block, 0, stream>>>(q, k, v, rk, rv, KT, VW, out);
    } else {
        rpa_flash_kernel<false><<<grid, block, 0, stream>>>(q, k, v, rk, rv,
                                                            nullptr, nullptr, out);
    }
}
